// CrossAttention_62371515072726
// MI455X (gfx1250) — matrix operand order repaired and checked by reading, not compiled, not run on hardware
//
#include <hip/hip_runtime.h>
#include <hip/hip_bf16.h>
#include <math.h>

// ---------------------------------------------------------------------------
// CDNA5 (gfx1250) cross-attention, all GEMM work on v_wmma_f32_16x16x32_bf16.
// cvt(x,ctx)->bf16; transpose+cvt weights; K/Q proj row-major, V proj written
// TRANSPOSED (so attention needs no LDS transpose); fused masked attention;
// output projection -> f32.
// ---------------------------------------------------------------------------

typedef __attribute__((ext_vector_type(16))) __bf16 v16bf;
typedef __attribute__((ext_vector_type(8)))  float  v8f;

static __device__ __forceinline__ v8f wmma_bf16(v16bf a, v16bf b, v8f c) {
    return __builtin_amdgcn_wmma_f32_16x16x32_bf16(false, a, false, b, (short)0, c, false, false);
}

union FragBF {
    v16bf v;
    uint4 u[2];
};

static __device__ __forceinline__ v8f zero8() {
    v8f z = {0.f, 0.f, 0.f, 0.f, 0.f, 0.f, 0.f, 0.f};
    return z;
}

#define BATCH   8
#define NSEQ    4096
#define MSEQ    77
#define MPAD    80      // M padded for 16-row score tiles
#define MPAD2   96      // M padded to multiple of 32 (WMMA K) for PV
#define EMB     1024
#define CTX     768
#define HEADS   16
#define HDIM    64

// ---------------------------------------------------------------------------
// Mask prep: additive mask (0 or -inf) for padded M
// ---------------------------------------------------------------------------
__global__ void mask_prep_kernel(const unsigned char* __restrict__ cmask,
                                 float* __restrict__ maskAdd) {
    int i = blockIdx.x * blockDim.x + threadIdx.x;
    if (i < BATCH * MPAD) {
        int b = i / MPAD;
        int m = i % MPAD;
        bool valid = (m < MSEQ) && (cmask[b * MSEQ + m] != 0);
        maskAdd[i] = valid ? 0.f : -__builtin_inff();
    }
}

// ---------------------------------------------------------------------------
// Zero-fill bf16 buffer (uint4 granularity: 8 halves / thread)
// ---------------------------------------------------------------------------
__global__ __launch_bounds__(256)
void zero_bf16_kernel(__bf16* __restrict__ p, long n8) {
    long i = blockIdx.x * 256L + threadIdx.x;
    if (i < n8) {
        uint4 z = {0u, 0u, 0u, 0u};
        ((uint4*)p)[i] = z;
    }
}

// ---------------------------------------------------------------------------
// f32 -> bf16 bulk convert (4 elements / thread)
// ---------------------------------------------------------------------------
struct bf4 { __bf16 a, b, c, d; };

__global__ __launch_bounds__(256)
void cvt_f32_bf16_kernel(const float* __restrict__ in, __bf16* __restrict__ out, long n4) {
    long i = blockIdx.x * 256L + threadIdx.x;
    if (i < n4) {
        float4 v = ((const float4*)in)[i];
        bf4 o;
        o.a = (__bf16)v.x; o.b = (__bf16)v.y; o.c = (__bf16)v.z; o.d = (__bf16)v.w;
        ((bf4*)out)[i] = o;
    }
}

// ---------------------------------------------------------------------------
// Weight transpose + convert: W[K][1024] f32 -> Wt[1024][K] bf16
// ---------------------------------------------------------------------------
__global__ __launch_bounds__(256)
void transpose_cvt_kernel(const float* __restrict__ W, __bf16* __restrict__ Wt, int K) {
    __shared__ float tile[32][33];
    const int tx = threadIdx.x & 31;
    const int ty = threadIdx.x >> 5;
    const int c0 = blockIdx.x * 32;
    const int k0 = blockIdx.y * 32;
#pragma unroll
    for (int j = 0; j < 4; ++j) {
        int kk = ty + j * 8;
        tile[kk][tx] = W[(size_t)(k0 + kk) * EMB + c0 + tx];
    }
    __syncthreads();
#pragma unroll
    for (int j = 0; j < 4; ++j) {
        int cc = ty + j * 8;
        Wt[(size_t)(c0 + cc) * K + k0 + tx] = (__bf16)tile[tx][cc];
    }
}

// ---------------------------------------------------------------------------
// bf16 WMMA GEMM: val = (sum_k A[r,k]*Wt[c,k] + bias[c]) * scale
//   OUT_MODE 0: f32 row-major         out[r*ldo + c]
//   OUT_MODE 1: bf16 row-major        out[r*ldo + c]
//   OUT_MODE 2: bf16 transposed/packed out[c*ldo + r]  (8 rows per b128 store)
//   128x128 tile / 256 threads (8 waves x 32x64), K-step 32,
//   double-buffered LDS, one barrier per K-step.
// ---------------------------------------------------------------------------
template <int OUT_MODE>
__global__ __launch_bounds__(256)
void gemm_bf16_wmma(const __bf16* __restrict__ A, long aBatch, int lda,
                    int rowsValid, int rowsOut, int Kdim,
                    const __bf16* __restrict__ Wt,
                    const float* __restrict__ bias,
                    float scale,
                    void* __restrict__ outv, long oBatch, int ldo) {
    __shared__ alignas(16) __bf16 lA[2][128][40];
    __shared__ alignas(16) __bf16 lB[2][128][40];

    const int tid  = threadIdx.x;
    const int lane = tid & 31;
    const int wave = tid >> 5;
    const int hi   = lane >> 4;
    const int ln   = lane & 15;
    const int rg   = wave >> 1;      // 0..3 -> 32-row group
    const int cg   = wave & 1;       // 0..1 -> 64-col group

    const int c0 = blockIdx.x * 128;
    const int r0 = blockIdx.y * 128;
    A += (size_t)blockIdx.z * aBatch;

    v8f acc[2][4];
#pragma unroll
    for (int fr = 0; fr < 2; ++fr)
#pragma unroll
        for (int t = 0; t < 4; ++t) acc[fr][t] = zero8();

    uint4 aReg[2], bReg[2];

    auto loadRegs = [&](int ks) {
#pragma unroll
        for (int j = 0; j < 2; ++j) {
            int u   = j * 256 + tid;     // 0..511
            int row = u >> 2;            // 0..127
            int seg = (u & 3) * 8;
            uint4 z = {0u, 0u, 0u, 0u};
            int gr = r0 + row;
            aReg[j] = z;
            if (gr < rowsValid)
                aReg[j] = *(const uint4*)(A + (size_t)gr * lda + ks * 32 + seg);
            bReg[j] = *(const uint4*)(Wt + (size_t)(c0 + row) * Kdim + ks * 32 + seg);
        }
    };
    auto storeLDS = [&](int buf) {
#pragma unroll
        for (int j = 0; j < 2; ++j) {
            int u   = j * 256 + tid;
            int row = u >> 2;
            int seg = (u & 3) * 8;
            *(uint4*)&lA[buf][row][seg] = aReg[j];
            *(uint4*)&lB[buf][row][seg] = bReg[j];
        }
    };
    auto compute = [&](int buf) {
        FragBF af[2];
#pragma unroll
        for (int fr = 0; fr < 2; ++fr) {
            const __bf16* pa = &lA[buf][rg * 32 + fr * 16 + ln][hi * 8];
            af[fr].u[0] = *(const uint4*)pa;
            af[fr].u[1] = *(const uint4*)(pa + 16);
        }
#pragma unroll
        for (int t = 0; t < 4; ++t) {
            FragBF bf;
            const __bf16* pb = &lB[buf][cg * 64 + t * 16 + ln][hi * 8];
            bf.u[0] = *(const uint4*)pb;
            bf.u[1] = *(const uint4*)(pb + 16);
#pragma unroll
            for (int fr = 0; fr < 2; ++fr)
                acc[fr][t] = wmma_bf16(af[fr].v, bf.v, acc[fr][t]);
        }
    };

    const int nk = Kdim / 32;
    loadRegs(0);
    storeLDS(0);
    for (int ks = 0; ks < nk; ++ks) {
        __syncthreads();
        if (ks + 1 < nk) loadRegs(ks + 1);
        compute(ks & 1);
        if (ks + 1 < nk) storeLDS((ks + 1) & 1);
    }

    // D mapping: row = i + 8*hi, col = ln within each 16x16 tile
#pragma unroll
    for (int fr = 0; fr < 2; ++fr)
#pragma unroll
        for (int t = 0; t < 4; ++t) {
            int c = c0 + cg * 64 + t * 16 + ln;
            float bv = bias[c];
            if (OUT_MODE == 2) {
                int rbase = r0 + rg * 32 + fr * 16 + hi * 8;
                if (rbase < rowsOut) {
                    union { uint4 u; __bf16 hh[8]; } pk;
#pragma unroll
                    for (int i = 0; i < 8; ++i)
                        pk.hh[i] = (__bf16)((acc[fr][t][i] + bv) * scale);
                    *(uint4*)&((__bf16*)outv)[(size_t)blockIdx.z * oBatch
                                              + (size_t)c * ldo + rbase] = pk.u;
                }
            } else {
#pragma unroll
                for (int i = 0; i < 8; ++i) {
                    int r = r0 + rg * 32 + fr * 16 + hi * 8 + i;
                    if (r < rowsOut) {
                        float val = (acc[fr][t][i] + bv) * scale;
                        if (OUT_MODE == 1)
                            ((__bf16*)outv)[(size_t)blockIdx.z * oBatch + (size_t)r * ldo + c] = (__bf16)val;
                        else
                            ((float*)outv)[(size_t)blockIdx.z * oBatch + (size_t)r * ldo + c] = val;
                    }
                }
            }
        }
}

// ---------------------------------------------------------------------------
// Fused attention: one wave per (b, h, 16-row n-tile).
//   S = Qtile(16x64) @ K_h^T(64x80)   2x5 WMMAs  (scale folded into Q)
//   masked softmax over 80; P padded to K=96 in LDS
//   O = P(16x96) @ V_h(96x64)         3x4 WMMAs, V frags direct from global Vt
// ---------------------------------------------------------------------------
__global__ __launch_bounds__(32)
void attention_kernel(const __bf16* __restrict__ Qbf,   // (B*N, 1024) pre-scaled
                      const __bf16* __restrict__ Kbf,   // (B, MPAD, 1024) row-major
                      const __bf16* __restrict__ Vt,    // (B, 1024, MPAD2) TRANSPOSED
                      const float*  __restrict__ maskAdd,
                      __bf16* __restrict__ attnOut) {
    __shared__ alignas(16) float  sS[16][MPAD];
    __shared__ alignas(16) __bf16 sP[16][104];

    const int lane = threadIdx.x & 31;
    const int hi   = lane >> 4;
    const int ln   = lane & 15;

    const int n0 = blockIdx.x * 16;
    const int h  = blockIdx.y;
    const int b  = blockIdx.z;

    // ---- Q fragments (2 K-chunks of 32 over d=64) ----
    FragBF qa[2];
#pragma unroll
    for (int c = 0; c < 2; ++c) {
        const __bf16* qrow = Qbf + ((size_t)(b * NSEQ + n0 + ln)) * EMB + h * HDIM + c * 32;
        qa[c].u[0] = *(const uint4*)(qrow + hi * 8);
        qa[c].u[1] = *(const uint4*)(qrow + 16 + hi * 8);
    }

    // ---- scores ----
    v8f s[5];
#pragma unroll
    for (int mt = 0; mt < 5; ++mt) s[mt] = zero8();
#pragma unroll
    for (int c = 0; c < 2; ++c) {
#pragma unroll
        for (int mt = 0; mt < 5; ++mt) {
            FragBF kb;   // B[k=d][col=m]: row-major K gives consecutive d
            const __bf16* krow = Kbf + ((size_t)(b * MPAD + mt * 16 + ln)) * EMB
                                 + h * HDIM + c * 32 + hi * 8;
            kb.u[0] = *(const uint4*)krow;
            kb.u[1] = *(const uint4*)(krow + 16);
            s[mt] = wmma_bf16(qa[c].v, kb.v, s[mt]);
        }
    }

#pragma unroll
    for (int mt = 0; mt < 5; ++mt)
#pragma unroll
        for (int i = 0; i < 8; ++i)
            sS[hi * 8 + i][mt * 16 + ln] = s[mt][i];
    __syncthreads();

    // ---- masked softmax, one row per lane (lanes 0..15) ----
    if (lane < 16) {
        const int r = lane;
        const float* ma = maskAdd + b * MPAD;
        float mx = -__builtin_inff();
        for (int m = 0; m < MPAD; ++m) {
            float v = sS[r][m] + ma[m];
            sS[r][m] = v;
            mx = fmaxf(mx, v);
        }
        float sum = 0.f;
        for (int m = 0; m < MPAD; ++m) {
            float e = __expf(sS[r][m] - mx);
            sS[r][m] = e;
            sum += e;
        }
        float inv = 1.f / sum;
        for (int m = 0; m < MPAD; ++m) sP[r][m] = (__bf16)(sS[r][m] * inv);
        for (int m = MPAD; m < MPAD2; ++m) sP[r][m] = (__bf16)0.f;
    }
    __syncthreads();

    // ---- O = P @ V_h : V B-frags straight from transposed global Vt ----
    v8f o[4];
#pragma unroll
    for (int t = 0; t < 4; ++t) o[t] = zero8();
#pragma unroll
    for (int c = 0; c < 3; ++c) {
        FragBF pa;
        const __bf16* pr = &sP[ln][c * 32 + hi * 8];
        pa.u[0] = *(const uint4*)pr;
        pa.u[1] = *(const uint4*)(pr + 16);
#pragma unroll
        for (int t = 0; t < 4; ++t) {
            FragBF vb;   // B[k=m][col=d]: Vt row (d) gives consecutive m
            const __bf16* vr = Vt + ((size_t)b * EMB + h * HDIM + t * 16 + ln) * MPAD2
                               + c * 32 + hi * 8;
            vb.u[0] = *(const uint4*)vr;
            vb.u[1] = *(const uint4*)(vr + 16);
            o[t] = wmma_bf16(pa.v, vb.v, o[t]);
        }
    }

#pragma unroll
    for (int t = 0; t < 4; ++t)
#pragma unroll
        for (int i = 0; i < 8; ++i)
            attnOut[((size_t)(b * NSEQ + n0 + hi * 8 + i)) * EMB
                    + h * HDIM + t * 16 + ln] = (__bf16)o[t][i];
}

// ---------------------------------------------------------------------------
// Host-side orchestration
// ---------------------------------------------------------------------------
static inline size_t alignUp(size_t x) { return (x + 255) & ~(size_t)255; }

extern "C" void kernel_launch(void* const* d_in, const int* in_sizes, int n_in,
                              void* d_out, int out_size, void* d_ws, size_t ws_size,
                              hipStream_t stream) {
    (void)in_sizes; (void)n_in; (void)out_size; (void)ws_size;

    const float*         x     = (const float*)d_in[0];
    const float*         ctx   = (const float*)d_in[1];
    const unsigned char* cmask = (const unsigned char*)d_in[2];
    const float* Wq = (const float*)d_in[3];
    const float* bq = (const float*)d_in[4];
    const float* Wk = (const float*)d_in[5];
    const float* bk = (const float*)d_in[6];
    const float* Wv = (const float*)d_in[7];
    const float* bv = (const float*)d_in[8];
    const float* Wo = (const float*)d_in[9];
    const float* bo = (const float*)d_in[10];
    float* out = (float*)d_out;

    const size_t ROWS = (size_t)BATCH * NSEQ;                 // 32768
    char* ws = (char*)d_ws;
    size_t off = 0;
    __bf16* Qbf   = (__bf16*)(ws + off); off += alignUp(ROWS * EMB * 2);
    __bf16* Xbf   = (__bf16*)(ws + off); off += alignUp(ROWS * EMB * 2);
    __bf16* Attn  = Xbf;  // x is dead after Q-projection; alias
    __bf16* Kbf   = (__bf16*)(ws + off); off += alignUp((size_t)BATCH * MPAD * EMB * 2);
    __bf16* Vtb   = (__bf16*)(ws + off); off += alignUp((size_t)BATCH * EMB * MPAD2 * 2);
    __bf16* Cbf   = (__bf16*)(ws + off); off += alignUp((size_t)BATCH * MSEQ * CTX * 2);
    __bf16* Wqt   = (__bf16*)(ws + off); off += alignUp((size_t)EMB * EMB * 2);
    __bf16* Wkt   = (__bf16*)(ws + off); off += alignUp((size_t)EMB * CTX * 2);
    __bf16* Wvt   = (__bf16*)(ws + off); off += alignUp((size_t)EMB * CTX * 2);
    __bf16* Wot   = (__bf16*)(ws + off); off += alignUp((size_t)EMB * EMB * 2);
    float*  maskA = (float*) (ws + off); off += alignUp((size_t)BATCH * MPAD * 4);

    const float qscale = 0.125f;   // 1/sqrt(64)

    // 1) prep
    mask_prep_kernel<<<dim3((BATCH * MPAD + 255) / 256), dim3(256), 0, stream>>>(cmask, maskA);
    {
        long n8 = (long)((size_t)BATCH * EMB * MPAD2 / 8);
        zero_bf16_kernel<<<dim3((unsigned)((n8 + 255) / 256)), dim3(256), 0, stream>>>(Vtb, n8);
    }
    {
        long n4 = (long)(ROWS * EMB / 4);
        cvt_f32_bf16_kernel<<<dim3((unsigned)((n4 + 255) / 256)), dim3(256), 0, stream>>>(x, Xbf, n4);
    }
    {
        long n4 = (long)((size_t)BATCH * MSEQ * CTX / 4);
        cvt_f32_bf16_kernel<<<dim3((unsigned)((n4 + 255) / 256)), dim3(256), 0, stream>>>(ctx, Cbf, n4);
    }
    transpose_cvt_kernel<<<dim3(EMB / 32, EMB / 32), dim3(256), 0, stream>>>(Wq, Wqt, EMB);
    transpose_cvt_kernel<<<dim3(EMB / 32, CTX / 32), dim3(256), 0, stream>>>(Wk, Wkt, CTX);
    transpose_cvt_kernel<<<dim3(EMB / 32, CTX / 32), dim3(256), 0, stream>>>(Wv, Wvt, CTX);
    transpose_cvt_kernel<<<dim3(EMB / 32, EMB / 32), dim3(256), 0, stream>>>(Wo, Wot, EMB);

    // 2) K projection (row-major out), V projection (transposed/packed out)
    gemm_bf16_wmma<1><<<dim3(EMB / 128, 1, BATCH), dim3(256), 0, stream>>>(
        Cbf, (long)MSEQ * CTX, CTX, MSEQ, MPAD, CTX, Wkt, bk, 1.f,
        (void*)Kbf, (long)MPAD * EMB, EMB);
    gemm_bf16_wmma<2><<<dim3(EMB / 128, 1, BATCH), dim3(256), 0, stream>>>(
        Cbf, (long)MSEQ * CTX, CTX, MSEQ, MPAD, CTX, Wvt, bv, 1.f,
        (void*)Vtb, (long)EMB * MPAD2, MPAD2);

    // 3) Q projection (scale folded)
    gemm_bf16_wmma<1><<<dim3(EMB / 128, (int)(ROWS / 128), 1), dim3(256), 0, stream>>>(
        Xbf, 0L, EMB, (int)ROWS, (int)ROWS, EMB, Wqt, bq, qscale,
        (void*)Qbf, 0L, EMB);

    // 4) fused attention -> Attn (bf16, aliases Xbf)
    attention_kernel<<<dim3(NSEQ / 16, HEADS, BATCH), dim3(32), 0, stream>>>(
        Qbf, Kbf, Vtb, maskA, Attn);

    // 5) output projection + bias -> f32 out
    gemm_bf16_wmma<0><<<dim3(EMB / 128, (int)(ROWS / 128), 1), dim3(256), 0, stream>>>(
        Attn, 0L, EMB, (int)ROWS, (int)ROWS, EMB, Wot, bo, 1.f,
        (void*)out, 0L, EMB);
}
